// CoordAttention_29824252903434
// MI455X (gfx1250) — matrix operand order repaired and checked by reading, not compiled, not run on hardware
//
#include <hip/hip_runtime.h>
#include <cfloat>
#include <cstdint>

// ---------------------------------------------------------------------------
// CDNA5 / gfx1250 CoordAttention.  wave32 WMMA bf16 pipeline with TDM
// (tensor_load_to_lds) staging of block-shared tiles:
//   convert -> GEMM(Q,K,V) -> flash attention (online softmax) -> GEMM(out)
// ---------------------------------------------------------------------------

typedef __attribute__((ext_vector_type(16))) __bf16        v16bf;
typedef __attribute__((ext_vector_type(8)))  float         v8f;
typedef __attribute__((ext_vector_type(4)))  unsigned int  u32x4;
typedef __attribute__((ext_vector_type(8)))  int           i32x8;
typedef __attribute__((ext_vector_type(4)))  int           i32x4;

union Frag16 { v16bf v; unsigned int u[8]; };
union FragC  { v8f   v; float        f[8]; };

// ---- WMMA fragment loaders (layouts per CDNA5 ISA 7.12.2, wave32) ----------
// A matrix 16x32 bf16: lane<16 -> row=lane, K = {0..7},{16..23}
//                      lane>=16 -> row=lane-16, K pattern +8 / +24
__device__ __forceinline__ v16bf load_frag_a(const __bf16* p, int ld, int lane) {
  const int row = lane & 15;
  const int kb  = (lane & 16) ? 8 : 0;
  const __bf16* rp = p + (size_t)row * (size_t)ld;
  Frag16 f;
#pragma unroll
  for (int i = 0; i < 8; ++i) {
    const int k = kb + ((i < 4) ? (2 * i) : (16 + 2 * (i - 4)));
    f.u[i] = *(const unsigned int*)(rp + k);   // packed (k, k+1), dword-aligned
  }
  return f.v;
}

// B matrix 32x16 bf16: col n = lane&15; lanes 0-15 hold K={0..7},{16..23},
// lanes 16-31 hold K={8..15},{24..31}.  Source must be k-contiguous per column, stride col_stride.
__device__ __forceinline__ v16bf load_frag_b(const __bf16* p, int col_stride, int lane) {
  const int n  = lane & 15;
  const int kb = (lane & 16) ? 8 : 0;
  const __bf16* cp = p + (size_t)n * (size_t)col_stride + kb;
  Frag16 f;
#pragma unroll
  for (int i = 0; i < 8; ++i) f.u[i] = *(const unsigned int*)(cp + ((i < 4) ? (2 * i) : (16 + 2 * (i - 4))));
  return f.v;
}

__device__ __forceinline__ v8f wmma_bf16(v16bf a, v16bf b, v8f c) {
  return __builtin_amdgcn_wmma_f32_16x16x32_bf16(false, a, false, b, (short)0, c, false, false);
}

// ---- TDM: 2D tile (dim0 contiguous, bf16) global -> LDS --------------------
// D# per CDNA5 ISA 8.3/8.4.  dim0/dim1 = tile AND tensor extents (in-bounds),
// stride0 in elements.  LDS result: [dim1][dim0] contiguous.
// This toolchain exposes the 6-arg builtin:
//   (u32x4 g0, i32x8 g1, i32x4 g2, i32x4 g3, i32x8 extra, i32 cpol)
__device__ __forceinline__ void tdm_load_2d(const __bf16* gsrc, unsigned lds_off,
                                            unsigned dim0, unsigned dim1,
                                            unsigned stride0) {
  const uint64_t ga = (uint64_t)(uintptr_t)gsrc;
  u32x4 g0;
  g0.x = 1u;                                                   // count=1 (valid)
  g0.y = lds_off;                                              // lds_addr
  g0.z = (unsigned)ga;                                         // global_addr lo
  g0.w = (unsigned)((ga >> 32) & 0x01FFFFFFu) | 0x80000000u;   // ga hi | type=2
  i32x8 g1;
  g1[0] = 0x00010000;                              // data_size = 1 (2 bytes)
  g1[1] = (int)((dim0 & 0xFFFFu) << 16);           // tensor_dim0 lo16
  g1[2] = (int)((dim0 >> 16) | ((dim1 & 0xFFFFu) << 16));   // d0 hi | d1 lo
  g1[3] = (int)((dim1 >> 16) | (dim0 << 16));      // d1 hi | tile_dim0
  g1[4] = (int)dim1;                               // tile_dim1 | tile_dim2=0
  g1[5] = (int)stride0;                            // tensor_dim0_stride lo32
  g1[6] = 0;                                       // stride0 hi | stride1 lo
  g1[7] = 0;
  i32x4 gz = {0, 0, 0, 0};
  i32x8 gz8 = {0, 0, 0, 0, 0, 0, 0, 0};
  __builtin_amdgcn_tensor_load_to_lds(g0, g1, gz, gz, gz8, 0);
}

// ---- conversion / packing kernels ------------------------------------------
__global__ void k_cvt(const float* __restrict__ s, __bf16* __restrict__ d, int n) {
  int i = blockIdx.x * 256 + threadIdx.x;
  if (i < n) d[i] = (__bf16)s[i];
}

// dst [8192][544] = concat(y[8192][512], coord_y[8192][3], zeros[29]) as bf16
__global__ void k_pack_yc(const float* __restrict__ y, const float* __restrict__ cy,
                          __bf16* __restrict__ d) {
  int i = blockIdx.x * 256 + threadIdx.x;
  if (i >= 8192 * 544) return;
  int row = i / 544, col = i - row * 544;
  float v = 0.f;
  if (col < 512)       v = y[(size_t)row * 512 + col];
  else if (col < 515)  v = cy[(size_t)row * 3 + (col - 512)];
  d[i] = (__bf16)v;
}

// W[K][N] (row-major, fp32)  ->  Wt[N][Kpad] bf16 (k-contiguous, zero padded)
__global__ void k_wt(const float* __restrict__ W, __bf16* __restrict__ Wt,
                     int K, int N, int Kpad) {
  int i = blockIdx.x * 256 + threadIdx.x;
  if (i >= N * Kpad) return;
  int n = i / Kpad, k = i - n * Kpad;
  Wt[i] = (__bf16)((k < K) ? W[(size_t)k * N + n] : 0.f);
}

// ---- generic WMMA GEMM: C[M][Nout] = A[M][K] * Wt[Nout][K]^T ---------------
// grid = (M/128, Nout/64), block = 128 (4 waves); each wave: 32x64 tile.
// Weight tile (64 cols x 32 k = 4KB) TDM-staged to LDS, double buffered.
// out_mode: 0 = bf16 row-major, 1 = bf16 per-batch transposed [b][Nout][rows],
//           2 = f32 row-major.
__global__ __launch_bounds__(128) void k_gemm(const __bf16* __restrict__ A,
                                              const __bf16* __restrict__ Wt,
                                              void* __restrict__ Cout,
                                              int K, int Nout, int out_mode,
                                              int batch_rows) {
  __shared__ __align__(16) __bf16 Bt[2][64 * 32];   // [buf][col][k]
  const int lane = threadIdx.x & 31;
  const int wave = threadIdx.x >> 5;
  const int m0 = (blockIdx.x * 4 + wave) * 32;
  const int n0 = blockIdx.y * 64;

  FragC acc[2][4];
#pragma unroll
  for (int rb = 0; rb < 2; ++rb)
#pragma unroll
    for (int t = 0; t < 4; ++t)
#pragma unroll
      for (int e = 0; e < 8; ++e) acc[rb][t].f[e] = 0.f;

  const int nsteps = K >> 5;
  if (wave == 0)
    tdm_load_2d(Wt + (size_t)n0 * K, (unsigned)(uintptr_t)&Bt[0][0], 32, 64, K);

  for (int s = 0; s < nsteps; ++s) {
    const int k0 = s << 5;
    if (wave == 0) __builtin_amdgcn_s_wait_tensorcnt(0);
    __syncthreads();                                   // buffer s&1 ready
    if (wave == 0 && s + 1 < nsteps)
      tdm_load_2d(Wt + (size_t)n0 * K + k0 + 32,
                  (unsigned)(uintptr_t)&Bt[(s + 1) & 1][0], 32, 64, K);

    const __bf16* bp = &Bt[s & 1][0];
    v16bf a0 = load_frag_a(A + (size_t)m0 * K + k0, K, lane);
    v16bf a1 = load_frag_a(A + (size_t)(m0 + 16) * K + k0, K, lane);
#pragma unroll
    for (int t = 0; t < 4; ++t) {
      v16bf b = load_frag_b(bp + (size_t)(16 * t) * 32, 32, lane);
      acc[0][t].v = wmma_bf16(a0, b, acc[0][t].v);
      acc[1][t].v = wmma_bf16(a1, b, acc[1][t].v);
    }
  }

  // C layout: VGPR j, lane<16 -> row j; lane>=16 -> row j+8; col = lane&15
  const int rowoff = (lane & 16) ? 8 : 0;
  const int cn = lane & 15;
#pragma unroll
  for (int rb = 0; rb < 2; ++rb)
#pragma unroll
    for (int t = 0; t < 4; ++t)
#pragma unroll
      for (int j = 0; j < 8; ++j) {
        const int r = m0 + 16 * rb + j + rowoff;
        const int c = n0 + 16 * t + cn;
        const float v = acc[rb][t].f[j];
        if (out_mode == 0) {
          ((__bf16*)Cout)[(size_t)r * Nout + c] = (__bf16)v;
        } else if (out_mode == 1) {
          const int bb = r / batch_rows, mm = r - bb * batch_rows;
          ((__bf16*)Cout)[((size_t)bb * Nout + c) * batch_rows + mm] = (__bf16)v;
        } else {
          ((float*)Cout)[(size_t)r * Nout + c] = v;
        }
      }
}

// ---- flash attention --------------------------------------------------------
// grid = (N/64, B*9), block = 128 (4 waves, one 16-row tile each; all waves of
// a block share (b,h) so K/V tiles are TDM-staged to LDS, double buffered).
// h<8: S = (Q Kt)*scale via WMMA; h==8: S = coord dot3 * coord_scale (VALU).
// Online softmax; P staged bf16 in wave-private LDS; P*V via WMMA.
__global__ __launch_bounds__(128) void k_flash(const __bf16* __restrict__ Qb,
                                               const __bf16* __restrict__ Kb,
                                               const __bf16* __restrict__ Vt,
                                               const float* __restrict__ cx,
                                               const float* __restrict__ cy,
                                               const unsigned char* __restrict__ mask,
                                               const float* __restrict__ cs,
                                               __bf16* __restrict__ Ab) {
  __shared__ __align__(16) __bf16 Pl[4][16 * 40];   // wave-private P tiles
  __shared__ __align__(16) __bf16 Kt[2][32 * 64];   // [buf][m][d]  (4KB)
  __shared__ __align__(16) __bf16 Vl[2][64 * 32];   // [buf][d][m]  (4KB)
  const int lane = threadIdx.x & 31;
  const int wave = threadIdx.x >> 5;
  const int bh = blockIdx.y;
  const int b = bh / 9;
  const int h = bh - b * 9;
  const int n0 = blockIdx.x * 64 + wave * 16;
  const bool coordh = (h == 8);                     // block-uniform
  const int rowoff = (lane & 16) ? 8 : 0;
  const int cn = lane & 15;
  const float cscale = cs[0];

  Frag16 aq0, aq1;
#pragma unroll
  for (int i = 0; i < 8; ++i) { aq0.u[i] = 0u; aq1.u[i] = 0u; }
  float cxr[8][3];
#pragma unroll
  for (int j = 0; j < 8; ++j) { cxr[j][0] = cxr[j][1] = cxr[j][2] = 0.f; }

  if (!coordh) {
    const __bf16* qp = Qb + ((size_t)(b * 2048 + n0)) * 512 + h * 64;
    aq0.v = load_frag_a(qp, 512, lane);
    aq1.v = load_frag_a(qp + 32, 512, lane);
  } else {
#pragma unroll
    for (int j = 0; j < 8; ++j) {
      const size_t r = (size_t)b * 2048 + n0 + j + rowoff;
      cxr[j][0] = cx[r * 3 + 0];
      cxr[j][1] = cx[r * 3 + 1];
      cxr[j][2] = cx[r * 3 + 2];
    }
  }

  float mrow[8], lrow[8];
  FragC acc[4];
#pragma unroll
  for (int j = 0; j < 8; ++j) { mrow[j] = -FLT_MAX; lrow[j] = 0.f; }
#pragma unroll
  for (int t = 0; t < 4; ++t)
#pragma unroll
    for (int e = 0; e < 8; ++e) acc[t].f[e] = 0.f;

  const __bf16* ksrc = Kb + ((size_t)b * 2048) * 512 + h * 64;   // + m*512
  const __bf16* vsrc = Vt + ((size_t)(b * 576 + h * 64)) * 2048; // + d*2048 + m

  if (wave == 0) {
    tdm_load_2d(vsrc, (unsigned)(uintptr_t)&Vl[0][0], 32, 64, 2048);
    if (!coordh)
      tdm_load_2d(ksrc, (unsigned)(uintptr_t)&Kt[0][0], 64, 32, 512);
  }

  for (int m0 = 0; m0 < 2048; m0 += 32) {
    const int buf = (m0 >> 5) & 1;
    if (wave == 0) __builtin_amdgcn_s_wait_tensorcnt(0);
    __syncthreads();                                 // K/V tiles for m0 ready
    if (wave == 0 && m0 + 32 < 2048) {
      tdm_load_2d(vsrc + (m0 + 32), (unsigned)(uintptr_t)&Vl[buf ^ 1][0],
                  32, 64, 2048);
      if (!coordh)
        tdm_load_2d(ksrc + (size_t)(m0 + 32) * 512,
                    (unsigned)(uintptr_t)&Kt[buf ^ 1][0], 64, 32, 512);
    }

    FragC s0, s1;
#pragma unroll
    for (int half = 0; half < 2; ++half) {
      FragC& s = half ? s1 : s0;
      const int mc = m0 + 16 * half;
      if (!coordh) {
        const __bf16* kp = &Kt[buf][0] + (size_t)(16 * half) * 64;
        v16bf bk0 = load_frag_b(kp, 64, lane);
        v16bf bk1 = load_frag_b(kp + 32, 64, lane);
        FragC c;
#pragma unroll
        for (int e = 0; e < 8; ++e) c.f[e] = 0.f;
        c.v = wmma_bf16(aq0.v, bk0, c.v);
        c.v = wmma_bf16(aq1.v, bk1, c.v);
#pragma unroll
        for (int j = 0; j < 8; ++j) s.f[j] = c.f[j] * 0.125f;   // 64^-0.5
      } else {
        const size_t mcol = (size_t)b * 2048 + mc + cn;
        const float c0 = cy[mcol * 3 + 0];
        const float c1 = cy[mcol * 3 + 1];
        const float c2 = cy[mcol * 3 + 2];
#pragma unroll
        for (int j = 0; j < 8; ++j)
          s.f[j] = (cxr[j][0] * c0 + cxr[j][1] * c1 + cxr[j][2] * c2) * cscale;
      }
      // mask (applied after scaling, matching reference)
      const unsigned char* mp =
          mask + ((size_t)(b * 2048 + n0 + rowoff)) * 2048 + mc + cn;
#pragma unroll
      for (int j = 0; j < 8; ++j)
        s.f[j] = mp[(size_t)j * 2048] ? s.f[j] : -FLT_MAX;
    }

    // tile row-max (16-lane reductions within each lane half) + online update
    float fac[8];
#pragma unroll
    for (int j = 0; j < 8; ++j) {
      float v = fmaxf(s0.f[j], s1.f[j]);
      v = fmaxf(v, __shfl_xor(v, 1, 16));
      v = fmaxf(v, __shfl_xor(v, 2, 16));
      v = fmaxf(v, __shfl_xor(v, 4, 16));
      v = fmaxf(v, __shfl_xor(v, 8, 16));
      const float mn = fmaxf(mrow[j], v);
      fac[j] = __expf(mrow[j] - mn);
      mrow[j] = mn;
    }
#pragma unroll
    for (int t = 0; t < 4; ++t)
#pragma unroll
      for (int j = 0; j < 8; ++j) acc[t].f[j] *= fac[j];

    // P = exp(S - m); bf16 into wave-private LDS; row-sum -> l
    __bf16* pw = &Pl[wave][0];
#pragma unroll
    for (int j = 0; j < 8; ++j) {
      const float p0 = __expf(s0.f[j] - mrow[j]);
      const float p1 = __expf(s1.f[j] - mrow[j]);
      pw[(size_t)(j + rowoff) * 40 + cn]      = (__bf16)p0;
      pw[(size_t)(j + rowoff) * 40 + 16 + cn] = (__bf16)p1;
      float t = p0 + p1;
      t += __shfl_xor(t, 1, 16);
      t += __shfl_xor(t, 2, 16);
      t += __shfl_xor(t, 4, 16);
      t += __shfl_xor(t, 8, 16);
      lrow[j] = lrow[j] * fac[j] + t;
    }

    // P (16x32) from LDS in A layout, then acc += P * V  (V from LDS tile)
    v16bf ap = load_frag_a(pw, 40, lane);
#pragma unroll
    for (int t = 0; t < 4; ++t) {
      v16bf bv = load_frag_b(&Vl[buf][0] + (size_t)(16 * t) * 32, 32, lane);
      acc[t].v = wmma_bf16(ap, bv, acc[t].v);
    }
  }

  // epilogue: O / l -> attn_out bf16 [8192][576]
#pragma unroll
  for (int t = 0; t < 4; ++t)
#pragma unroll
    for (int j = 0; j < 8; ++j) {
      const float o = acc[t].f[j] / lrow[j];
      Ab[((size_t)(b * 2048 + n0 + j + rowoff)) * 576 + h * 64 + 16 * t + cn] =
          (__bf16)o;
    }
}

// ---------------------------------------------------------------------------
extern "C" void kernel_launch(void* const* d_in, const int* in_sizes, int n_in,
                              void* d_out, int out_size, void* d_ws, size_t ws_size,
                              hipStream_t stream) {
  (void)in_sizes; (void)n_in; (void)out_size; (void)ws_size;
  const float* x   = (const float*)d_in[0];
  const float* y   = (const float*)d_in[1];
  const float* cx  = (const float*)d_in[2];
  const float* cy  = (const float*)d_in[3];
  const unsigned char* mask = (const unsigned char*)d_in[4];
  const float* Wq  = (const float*)d_in[5];
  const float* Wk  = (const float*)d_in[6];
  const float* Wv  = (const float*)d_in[7];
  const float* Wo  = (const float*)d_in[8];
  const float* cs  = (const float*)d_in[9];

  char* base = (char*)d_ws;
  size_t off = 0;
  auto take = [&](size_t elts) -> __bf16* {
    __bf16* p = (__bf16*)(base + off);
    off += (elts * 2 + 255) & ~(size_t)255;
    return p;
  };

  __bf16* Xb  = take((size_t)8192 * 512);
  __bf16* Yb  = take((size_t)8192 * 512);
  __bf16* YCb = take((size_t)8192 * 544);
  __bf16* Qb  = take((size_t)8192 * 512);
  __bf16* Kbf = take((size_t)8192 * 512);
  __bf16* Vt  = take((size_t)4 * 576 * 2048);
  __bf16* Ab  = take((size_t)8192 * 576);
  __bf16* Wqt = take((size_t)512 * 512);
  __bf16* Wkt = take((size_t)512 * 512);
  __bf16* Wvt = take((size_t)576 * 544);
  __bf16* Wot = take((size_t)512 * 576);

  const int n1 = 8192 * 512;
  k_cvt<<<(n1 + 255) / 256, 256, 0, stream>>>(x, Xb, n1);
  k_cvt<<<(n1 + 255) / 256, 256, 0, stream>>>(y, Yb, n1);
  const int n2 = 8192 * 544;
  k_pack_yc<<<(n2 + 255) / 256, 256, 0, stream>>>(y, cy, YCb);
  k_wt<<<(512 * 512 + 255) / 256, 256, 0, stream>>>(Wq, Wqt, 512, 512, 512);
  k_wt<<<(512 * 512 + 255) / 256, 256, 0, stream>>>(Wk, Wkt, 512, 512, 512);
  k_wt<<<(576 * 544 + 255) / 256, 256, 0, stream>>>(Wv, Wvt, 515, 576, 544);
  k_wt<<<(512 * 576 + 255) / 256, 256, 0, stream>>>(Wo, Wot, 576, 512, 576);

  // Q = Xb * Wq   (bf16 row-major [8192][512])
  k_gemm<<<dim3(64, 8), 128, 0, stream>>>(Xb, Wqt, Qb, 512, 512, 0, 0);
  // K = Yb * Wk
  k_gemm<<<dim3(64, 8), 128, 0, stream>>>(Yb, Wkt, Kbf, 512, 512, 0, 0);
  // V^T = (Yb|coord) * Wv, stored [b][576][2048] (m-contiguous for WMMA B)
  k_gemm<<<dim3(64, 9), 128, 0, stream>>>(YCb, Wvt, Vt, 544, 576, 1, 2048);

  // flash attention over all (b, h, 16-row tiles)
  k_flash<<<dim3(32, 36), 128, 0, stream>>>(Qb, Kbf, Vt, cx, cy, mask, cs, Ab);

  // out = Ab * Wo  (f32 row-major [8192][512] straight into d_out)
  k_gemm<<<dim3(64, 8), 128, 0, stream>>>(Ab, Wot, d_out, 576, 512, 2, 0);
}
